// Int16Conv1x1Q88_30983894073630
// MI455X (gfx1250) — 8-bit matrix operand order repaired and checked by reading, compiled, not run on hardware
//
#include <hip/hip_runtime.h>
#include <hip/hip_bf16.h>
#include <stdint.h>


typedef int v8i __attribute__((ext_vector_type(8)));

#define CIN   512
#define COUT  512
#define LDIM  4096
#define NBAT  8

#define WSTRIDE 72
#define XSTRIDE 80

__device__ __forceinline__ int q88(float v) {
    float r = rintf(v * 256.0f);
    r = fminf(fmaxf(r, -32768.0f), 32767.0f);
    return (int)r;
}

__device__ __forceinline__ v8i lds_load_a(const unsigned char* p) {
    union { unsigned long long u[4]; v8i v; } r;
    r.u[0] = *(const unsigned long long*)(p +  0);
    r.u[1] = *(const unsigned long long*)(p +  8);
    r.u[2] = *(const unsigned long long*)(p + 32);
    r.u[3] = *(const unsigned long long*)(p + 40);
    return r.v;
}

__device__ __forceinline__ v8i lds_load_b(const unsigned char* p) {
    union { int4 q[2]; v8i v; } r;
    r.q[0] = *(const int4*)(p +  0);
    r.q[1] = *(const int4*)(p + 32);
    return r.v;
}

__global__ __launch_bounds__(256)
void Int16Conv1x1Q88_kernel(const float* __restrict__ x,
                            const int16_t* __restrict__ wq,
                            const int16_t* __restrict__ bq,
                            float* __restrict__ out) {
    __shared__ __align__(16) unsigned char sWL[64 * WSTRIDE];
    __shared__ __align__(16) unsigned char sWH[64 * WSTRIDE];
    __shared__ __align__(16) unsigned char sXL[64 * XSTRIDE];
    __shared__ __align__(16) unsigned char sXH[64 * XSTRIDE];

    const int tid  = threadIdx.x;
    const int lane = tid & 31;
    const int wv   = tid >> 5;
    const int lt   = wv & 3;
    const int otb  = (wv >> 2) * 2;

    const int b  = blockIdx.z;
    const int o0 = blockIdx.y * 64;
    const int l0 = blockIdx.x * 64;

    const v8i vzero = {0, 0, 0, 0, 0, 0, 0, 0};
    v8i acc[2][4];
    #pragma unroll
    for (int t = 0; t < 2; ++t)
        #pragma unroll
        for (int pnl = 0; pnl < 4; ++pnl)
            acc[t][pnl] = vzero;

    const int mrow  = lane & 15;
    const int akoff = (lane < 16) ? 0 : 16;
    const int bkoff = (lane < 16) ? 0 : 16;

    #pragma unroll 1
    for (int kc = 0; kc < CIN / 64; ++kc) {
        const int c0 = kc * 64;

        #pragma unroll
        for (int it = 0; it < 2; ++it) {
            const int idx = it * 256 + tid;
            const int row = idx >> 3;
            const int cb  = idx & 7;
            const int4 v = *(const int4*)(wq + (size_t)(o0 + row) * CIN + c0 + cb * 8);
            const unsigned a0 = (unsigned)v.x, a1 = (unsigned)v.y,
                           a2 = (unsigned)v.z, a3 = (unsigned)v.w;
            const unsigned lo01 = (a0 & 255u) | (((a0 >> 16) & 255u) << 8) |
                                  ((a1 & 255u) << 16) | (((a1 >> 16) & 255u) << 24);
            const unsigned lo23 = (a2 & 255u) | (((a2 >> 16) & 255u) << 8) |
                                  ((a3 & 255u) << 16) | (((a3 >> 16) & 255u) << 24);
            const unsigned hi01 = ((a0 >> 8) & 255u) | (((a0 >> 24) & 255u) << 8) |
                                  (((a1 >> 8) & 255u) << 16) | (((a1 >> 24) & 255u) << 24);
            const unsigned hi23 = ((a2 >> 8) & 255u) | (((a2 >> 24) & 255u) << 8) |
                                  (((a3 >> 8) & 255u) << 16) | (((a3 >> 24) & 255u) << 24);
            unsigned* dL = (unsigned*)(sWL + row * WSTRIDE + cb * 8);
            unsigned* dH = (unsigned*)(sWH + row * WSTRIDE + cb * 8);
            dL[0] = lo01; dL[1] = lo23;
            dH[0] = hi01; dH[1] = hi23;
        }

        #pragma unroll
        for (int it = 0; it < 4; ++it) {
            const int idx = it * 256 + tid;
            const int ll  = idx & 63;
            const int cg  = idx >> 6;
            const float* xp = x + ((size_t)b * CIN + c0 + cg * 4) * LDIM + l0 + ll;
            const int q0 = q88(xp[0 * LDIM]);
            const int q1 = q88(xp[1 * LDIM]);
            const int q2 = q88(xp[2 * LDIM]);
            const int q3 = q88(xp[3 * LDIM]);
            const unsigned lo = (unsigned)(q0 & 255) | ((unsigned)(q1 & 255) << 8) |
                                ((unsigned)(q2 & 255) << 16) | ((unsigned)(q3 & 255) << 24);
            const unsigned hi = (unsigned)((q0 >> 8) & 255) | ((unsigned)((q1 >> 8) & 255) << 8) |
                                ((unsigned)((q2 >> 8) & 255) << 16) | ((unsigned)((q3 >> 8) & 255) << 24);
            *(unsigned*)(sXL + ll * XSTRIDE + cg * 4) = lo;
            *(unsigned*)(sXH + ll * XSTRIDE + cg * 4) = hi;
        }

        __syncthreads();

        {
            const v8i BH = lds_load_b(sXH + (lt * 16 + mrow) * XSTRIDE + bkoff);
            const v8i BL = lds_load_b(sXL + (lt * 16 + mrow) * XSTRIDE + bkoff);
            #pragma unroll
            for (int t = 0; t < 2; ++t) {
                const int ot = otb + t;
                const v8i AH = lds_load_a(sWH + (ot * 16 + mrow) * WSTRIDE + akoff);
                const v8i AL = lds_load_a(sWL + (ot * 16 + mrow) * WSTRIDE + akoff);
                acc[t][0] = __builtin_amdgcn_wmma_i32_16x16x64_iu8(true,  AH, true,  BH, acc[t][0], false, false);
                acc[t][1] = __builtin_amdgcn_wmma_i32_16x16x64_iu8(true,  AH, false, BL, acc[t][1], false, false);
                acc[t][2] = __builtin_amdgcn_wmma_i32_16x16x64_iu8(false, AL, true,  BH, acc[t][2], false, false);
                acc[t][3] = __builtin_amdgcn_wmma_i32_16x16x64_iu8(false, AL, false, BL, acc[t][3], false, false);
            }
        }

        __syncthreads();
    }

    const int n     = lane & 15;
    const int mbase = (lane < 16) ? 0 : 8;
    #pragma unroll
    for (int t = 0; t < 2; ++t) {
        const int ot = otb + t;
        #pragma unroll
        for (int i = 0; i < 8; ++i) {
            const int m = mbase + i;
            const unsigned acc32 = ((unsigned)acc[t][0][i] << 16) +
                                   ((unsigned)(acc[t][1][i] + acc[t][2][i]) << 8) +
                                   (unsigned)acc[t][3][i];
            const int a  = (int)acc32;
            const int o  = o0 + ot * 16 + m;
            int yq = ((a + 128) >> 8) + (int)bq[o];
            yq = yq < -32768 ? -32768 : (yq > 32767 ? 32767 : yq);
            out[((size_t)b * COUT + o) * LDIM + l0 + lt * 16 + n] = (float)yq * (1.0f / 256.0f);
        }
    }
}

extern "C" void kernel_launch(void* const* d_in, const int* in_sizes, int n_in,
                              void* d_out, int out_size, void* d_ws, size_t ws_size,
                              hipStream_t stream) {
    const float*   x  = (const float*)d_in[0];
    const int16_t* wq = (const int16_t*)d_in[1];
    const int16_t* bq = (const int16_t*)d_in[2];
    float* out = (float*)d_out;

    dim3 grid(LDIM / 64, COUT / 64, NBAT);
    dim3 block(256);
    Int16Conv1x1Q88_kernel<<<grid, block, 0, stream>>>(x, wq, bq, out);
}
